// BatchHardTripletLoss_30520037605879
// MI455X (gfx1250) — matrix operand order repaired and checked by reading, not compiled, not run on hardware
//
#include <hip/hip_runtime.h>
#include <hip/hip_bf16.h>

typedef _Float16 f16;
typedef __attribute__((ext_vector_type(16))) _Float16 v16h;
typedef __attribute__((ext_vector_type(8)))  _Float16 v8h;
typedef __attribute__((ext_vector_type(8)))  float    v8f;

#define BATCH 8192
#define DIM   128
#define NCHUNK 8
#define COLS_PER_CHUNK (BATCH / NCHUNK)   // 1024
#define WAVES_PER_BLOCK 4
#define TRIPLET_MARGIN 1.0f

// ---------------------------------------------------------------------------
// Kernel 1: f32 -> f16 conversion + per-row squared norms (f32).
// One 128-thread block per row; coalesced along D.
// ---------------------------------------------------------------------------
__global__ void __launch_bounds__(DIM)
bh_prep_kernel(const float* __restrict__ X, f16* __restrict__ Xh,
               float* __restrict__ sq) {
  const int row = blockIdx.x;
  const int tid = threadIdx.x;
  float v = X[(size_t)row * DIM + tid];
  Xh[(size_t)row * DIM + tid] = (f16)v;
  float s = v * v;
  #pragma unroll
  for (int m = 16; m > 0; m >>= 1) s += __shfl_xor(s, m, 32);
  __shared__ float part[DIM / 32];
  const int wave = tid >> 5, lane = tid & 31;
  if (lane == 0) part[wave] = s;
  __syncthreads();
  if (tid == 0) sq[row] = part[0] + part[1] + part[2] + part[3];
}

// ---------------------------------------------------------------------------
// Kernel 2: fused Gram-tile WMMA + masked squared-distance epilogue.
// One wave per 16-row anchor tile; blockIdx.y selects a 1024-column chunk.
// B fragments are register double-buffered so the WMMA chain never waits on
// its own loads; hardest-pos/neg tracked in d^2 space (sqrt is monotonic,
// deferred to the final reduction).
// ---------------------------------------------------------------------------
__global__ void __launch_bounds__(32 * WAVES_PER_BLOCK)
bh_tile_kernel(const f16* __restrict__ Xh, const float* __restrict__ sq,
               const int* __restrict__ labels,
               float* __restrict__ php, float* __restrict__ phn) {
  const int wave = threadIdx.x >> 5;
  const int lane = threadIdx.x & 31;
  const int half = lane >> 4;     // 0: lanes 0-15, 1: lanes 16-31
  const int lid  = lane & 15;

  const int row_base  = (blockIdx.x * WAVES_PER_BLOCK + wave) * 16;
  const int col_start = blockIdx.y * COLS_PER_CHUNK;
  const int NT        = COLS_PER_CHUNK / 16;

  // --- Preload A fragments (16 rows x K=128), ISA 16-bit A 16x32 layout:
  // lane<16 row M=lid holds K = {0..7, 16..23}; lane>=16 holds K = {8..15, 24..31}
  const int arow = row_base + lid;
  v16h afrag[4];
  #pragma unroll
  for (int kk = 0; kk < 4; ++kk) {
    const f16* ap = Xh + (size_t)arow * DIM + kk * 32 + half * 8;
    v8h alo = *(const v8h*)(ap);        // K = kk*32 + half*8 + 0..7
    v8h ahi = *(const v8h*)(ap + 16);   // K = kk*32 + 16 + half*8 + 0..7
    v16h a;
    #pragma unroll
    for (int i = 0; i < 8; ++i) { a[i] = alo[i]; a[8 + i] = ahi[i]; }
    afrag[kk] = a;
  }

  // Row-side metadata for the 8 C/D rows this lane accumulates
  // (VGPR r, this half -> global row = row_base + half*8 + r).
  float rsq[8]; int rlab[8];
  #pragma unroll
  for (int r = 0; r < 8; ++r) {
    const int rr = row_base + half * 8 + r;
    rsq[r]  = sq[rr];
    rlab[r] = labels[rr];
  }

  const float INF = __builtin_inff();
  float hp2[8], hn2[8];          // hardest pos (max) / neg (min) in d^2 space
  #pragma unroll
  for (int r = 0; r < 8; ++r) { hp2[r] = -INF; hn2[r] = INF; }

  // B 32x16 16-bit layout: lanes 0-15 hold K={0..7,16..23} of column N=lid,
  // lanes 16-31 hold K={8..15,24..31} (same order as A).
  auto load_tile = [&](int ct, v16h* bb, float& cs, int& cl) {
    const int col = col_start + ct * 16 + lid;
    #pragma unroll
    for (int kk = 0; kk < 4; ++kk) {
      const f16* bp = Xh + (size_t)col * DIM + kk * 32 + half * 8;
      v8h blo = *(const v8h*)(bp);
      v8h bhi = *(const v8h*)(bp + 16);
      v16h b;
      #pragma unroll
      for (int i = 0; i < 8; ++i) { b[i] = blo[i]; b[8 + i] = bhi[i]; }
      bb[kk] = b;
    }
    cs = sq[col];
    cl = labels[col];
  };

  // Prologue: prefetch tile 0.
  v16h bbuf[4]; float csq; int clab;
  load_tile(0, bbuf, csq, clab);

  for (int ct = 0; ct < NT; ++ct) {
    // Issue next tile's loads before the WMMA chain (uniform branch,
    // EXEC stays all-1s for WMMA).
    v16h bnxt[4]; float nsq = 0.0f; int nlab = 0;
    if (ct + 1 < NT) load_tile(ct + 1, bnxt, nsq, nlab);

    // WMMA chain consumes registers loaded last iteration: no load wait here.
    v8f acc = {};
    #pragma unroll
    for (int kk = 0; kk < 4; ++kk) {
      acc = __builtin_amdgcn_wmma_f32_16x16x32_f16(
          false, afrag[kk], false, bbuf[kk], (short)0, acc, false, false);
    }

    const int col = col_start + ct * 16 + lid;
    #pragma unroll
    for (int r = 0; r < 8; ++r) {
      const int rr = row_base + half * 8 + r;
      float d2 = fmaxf(rsq[r] + csq - 2.0f * acc[r], 0.0f);
      const bool same = (rlab[r] == clab);
      const bool diag = (rr == col);
      hp2[r] = fmaxf(hp2[r], (same && !diag) ? d2 : -INF);
      hn2[r] = fminf(hn2[r], same ? INF : d2);
    }

    #pragma unroll
    for (int kk = 0; kk < 4; ++kk) bbuf[kk] = bnxt[kk];
    csq = nsq; clab = nlab;
  }

  // Cross-lane fold over the 16 columns held by this half-group
  // (xor masks 8..1 stay within each 16-lane group of wave32).
  #pragma unroll
  for (int r = 0; r < 8; ++r) {
    #pragma unroll
    for (int m = 8; m > 0; m >>= 1) {
      hp2[r] = fmaxf(hp2[r], __shfl_xor(hp2[r], m, 32));
      hn2[r] = fminf(hn2[r], __shfl_xor(hn2[r], m, 32));
    }
  }
  if (lid == 0) {
    const int chunk = blockIdx.y;
    #pragma unroll
    for (int r = 0; r < 8; ++r) {
      const int rr = row_base + half * 8 + r;
      php[(size_t)chunk * BATCH + rr] = hp2[r];
      phn[(size_t)chunk * BATCH + rr] = hn2[r];
    }
  }
}

// ---------------------------------------------------------------------------
// Kernel 3: fold chunk partials per row (still d^2 space), take sqrt once,
// apply validity + relu, mean -> scalar.
// ---------------------------------------------------------------------------
__global__ void __launch_bounds__(1024)
bh_reduce_kernel(const float* __restrict__ php, const float* __restrict__ phn,
                 float* __restrict__ out) {
  const float INF = __builtin_inff();
  float sum = 0.0f, cnt = 0.0f;
  for (int row = threadIdx.x; row < BATCH; row += 1024) {
    float hp2 = -INF, hn2 = INF;
    #pragma unroll
    for (int c = 0; c < NCHUNK; ++c) {
      hp2 = fmaxf(hp2, php[(size_t)c * BATCH + row]);
      hn2 = fminf(hn2, phn[(size_t)c * BATCH + row]);
    }
    if (hp2 > -INF && hn2 < INF) {     // any positive AND any negative
      const float hp = sqrtf(hp2);     // sqrt deferred here: monotonic
      const float hn = sqrtf(hn2);
      sum += fmaxf(hp - hn + TRIPLET_MARGIN, 0.0f);
      cnt += 1.0f;
    }
  }
  #pragma unroll
  for (int m = 16; m > 0; m >>= 1) {
    sum += __shfl_xor(sum, m, 32);
    cnt += __shfl_xor(cnt, m, 32);
  }
  __shared__ float ssum[32], scnt[32];
  const int wave = threadIdx.x >> 5, lane = threadIdx.x & 31;
  if (lane == 0) { ssum[wave] = sum; scnt[wave] = cnt; }
  __syncthreads();
  if (threadIdx.x == 0) {
    float S = 0.0f, C = 0.0f;
    for (int w = 0; w < 32; ++w) { S += ssum[w]; C += scnt[w]; }
    out[0] = (C > 0.0f) ? (S / C) : 0.0f;
  }
}

// ---------------------------------------------------------------------------
extern "C" void kernel_launch(void* const* d_in, const int* in_sizes, int n_in,
                              void* d_out, int out_size, void* d_ws, size_t ws_size,
                              hipStream_t stream) {
  const float* X      = (const float*)d_in[0];
  const int*   labels = (const int*)d_in[1];
  float*       out    = (float*)d_out;

  // Workspace layout: Xh (2 MB) | sq (32 KB) | php (256 KB) | phn (256 KB)
  char*  ws  = (char*)d_ws;
  f16*   Xh  = (f16*)ws;
  float* sq  = (float*)(ws + (size_t)BATCH * DIM * sizeof(f16));
  float* php = (float*)(ws + (size_t)BATCH * DIM * sizeof(f16) + (size_t)BATCH * sizeof(float));
  float* phn = php + (size_t)NCHUNK * BATCH;

  bh_prep_kernel<<<BATCH, DIM, 0, stream>>>(X, Xh, sq);

  dim3 grid(BATCH / 16 / WAVES_PER_BLOCK, NCHUNK);   // (128, 8)
  bh_tile_kernel<<<grid, 32 * WAVES_PER_BLOCK, 0, stream>>>(Xh, sq, labels, php, phn);

  bh_reduce_kernel<<<1, 1024, 0, stream>>>(php, phn, out);
}
